// MultiHeadMaskedAttention_2765958938888
// MI455X (gfx1250) — matrix operand order repaired and checked by reading, not compiled, not run on hardware
//
#include <hip/hip_runtime.h>
#include <hip/hip_bf16.h>

// ---------------------------------------------------------------------------
// Problem constants (match reference)
// ---------------------------------------------------------------------------
#define BATCH 2
#define TSEQ  2048
#define CDIM  2048
#define PTLEN 1024
#define SSEQ  3072          // PTLEN + TSEQ
#define NH    16
#define DHEAD 128
#define N3C   6144          // 3*CDIM

typedef __attribute__((ext_vector_type(16))) __bf16 bf16x16;
typedef __attribute__((ext_vector_type(8)))  float  floatx8;

struct __align__(16) U4 { unsigned int x, y, z, w; };

__device__ __forceinline__ unsigned short f32_to_bf16(float f) {
    unsigned int u = __float_as_uint(f);
    u += 0x7FFFu + ((u >> 16) & 1u);       // round-to-nearest-even
    return (unsigned short)(u >> 16);
}

__device__ __forceinline__ floatx8 wmma_bf16(bf16x16 a, bf16x16 b, floatx8 c) {
    return __builtin_amdgcn_wmma_f32_16x16x32_bf16(
        false, a, false, b, (short)0, c, false, false);
}

// ---------------------------------------------------------------------------
// Kernel 1: copy past K/V into f32 output cache + bf16 mirrors in workspace
// ---------------------------------------------------------------------------
__global__ void copy_past(const float* __restrict__ pk, const float* __restrict__ pv,
                          float* __restrict__ outk, float* __restrict__ outv,
                          unsigned short* __restrict__ kbf, unsigned short* __restrict__ vbf)
{
    size_t i = (size_t)blockIdx.x * blockDim.x + threadIdx.x;  // over B*PT*C
    if (i >= (size_t)BATCH * PTLEN * CDIM) return;
    size_t b  = i / ((size_t)PTLEN * CDIM);
    size_t rc = i - b * ((size_t)PTLEN * CDIM);
    size_t dst = b * (size_t)SSEQ * CDIM + rc;
    float kv = pk[i], vv = pv[i];
    outk[dst] = kv;  outv[dst] = vv;
    kbf[dst] = f32_to_bf16(kv);
    vbf[dst] = f32_to_bf16(vv);
}

// ---------------------------------------------------------------------------
// Kernel 2: qkv = x @ W_qkv + b_qkv   (M=4096, N=6144, K=2048)
// 256 threads = 8 waves; 128x128 tile; each wave owns 16 rows x 128 cols.
// A/B staged to LDS pre-swizzled into WMMA fragment order.
// ---------------------------------------------------------------------------
__global__ __launch_bounds__(256)
void qkv_gemm(const float* __restrict__ x, const float* __restrict__ W,
              const float* __restrict__ bias,
              unsigned short* __restrict__ q_bf,
              float* __restrict__ out_k, float* __restrict__ out_v,
              unsigned short* __restrict__ k_bf, unsigned short* __restrict__ v_bf)
{
    __shared__ __align__(32) unsigned short ldsA[8 * 512]; // 8 M-subtiles, frag-linear
    __shared__ __align__(32) unsigned short ldsB[8 * 512]; // 8 N-subtiles, frag-linear

    const int t = threadIdx.x, wave = t >> 5, lane = t & 31;
    const int m0 = blockIdx.y * 128;
    const int n0 = blockIdx.x * 128;

    floatx8 acc[8] = {};

    for (int k0 = 0; k0 < CDIM; k0 += 32) {
        // ---- stage A (x, f32 -> bf16): thread covers (row m, 16 contiguous k)
        {
            int m = t >> 1, half = t & 1;
            const float* src = x + (size_t)(m0 + m) * CDIM + k0 + half * 16;
            if (k0 + 32 < CDIM) __builtin_prefetch(src + 32, 0, 0);
            float f[16];
#pragma unroll
            for (int i = 0; i < 16; i++) f[i] = src[i];
            int sub = m >> 4, lr = m & 15;
#pragma unroll
            for (int i = 0; i < 16; i += 2) {
                int ln   = lr + ((i & 8) ? 16 : 0);          // (k&8) selects lane half
                int slot = (i & 7) + (half ? 8 : 0);          // (k>=16) selects hi slots
                unsigned int p = (unsigned int)f32_to_bf16(f[i]) |
                                 ((unsigned int)f32_to_bf16(f[i + 1]) << 16);
                *(unsigned int*)&ldsA[sub * 512 + ln * 16 + slot] = p;
            }
        }
        // ---- stage B (W_qkv, f32 -> bf16): thread covers (col n, 16 contiguous k)
        {
            int n = t & 127, kh = t >> 7;
            const float* src = W + (size_t)(k0 + kh * 16) * N3C + n0 + n;
            unsigned short tmp[16];
#pragma unroll
            for (int i = 0; i < 16; i++) tmp[i] = f32_to_bf16(src[(size_t)i * N3C]);
            int sub = n >> 4;
            int ln  = (n & 15);
#pragma unroll
            for (int i = 0; i < 16; i += 2)
                *(unsigned int*)&ldsB[sub * 512 + (ln + ((i & 8) ? 16 : 0)) * 16 + (i & 7) + kh * 8] =
                    (unsigned int)tmp[i] | ((unsigned int)tmp[i + 1] << 16);
        }
        __syncthreads();

        bf16x16 af = *(const bf16x16*)&ldsA[wave * 512 + lane * 16];
#pragma unroll
        for (int j = 0; j < 8; j++) {
            bf16x16 bf = *(const bf16x16*)&ldsB[j * 512 + lane * 16];
            acc[j] = wmma_bf16(af, bf, acc[j]);
        }
        __syncthreads();
    }

    // ---- epilogue: bias + route to q (bf16 ws) / k,v (f32 out + bf16 ws)
    const int third = n0 / CDIM;        // uniform per block (128 | 2048)
    const int g = lane >> 4, nl = lane & 15;
#pragma unroll
    for (int j = 0; j < 8; j++) {
        int col3 = n0 + j * 16 + nl;
        float bv = bias[col3];
#pragma unroll
        for (int r = 0; r < 8; r++) {
            int row = m0 + wave * 16 + g * 8 + r;       // 0..4095  (= b*T + t)
            float val = acc[j][r] + bv;
            if (third == 0) {
                q_bf[(size_t)row * CDIM + col3] = f32_to_bf16(val);
            } else {
                int c = col3 - third * CDIM;
                int b = row >> 11, tt = row & (TSEQ - 1);
                size_t idx = ((size_t)b * SSEQ + (PTLEN + tt)) * CDIM + c;
                if (third == 1) { out_k[idx] = val; k_bf[idx] = f32_to_bf16(val); }
                else           { out_v[idx] = val; v_bf[idx] = f32_to_bf16(val); }
            }
        }
    }
}

// ---------------------------------------------------------------------------
// Kernel 3: flash attention. Block = (b, head, 64-query tile), 4 waves.
// Wave owns 16 query rows; loop over 64-key chunks with online softmax.
// ---------------------------------------------------------------------------
__global__ __launch_bounds__(128)
void attn(const unsigned short* __restrict__ q_bf,
          const unsigned short* __restrict__ k_bf,
          const unsigned short* __restrict__ v_bf,
          unsigned short* __restrict__ o_bf)
{
    const int qt = blockIdx.x, h = blockIdx.y, b = blockIdx.z;
    const int t = threadIdx.x, wave = t >> 5, lane = t & 31;
    const int g = lane >> 4, nl = lane & 15;

    __shared__ __align__(32) unsigned short ldsK[16 * 512]; // (dstep 0..3, keysub 0..3)
    __shared__ __align__(32) unsigned short ldsV[16 * 512]; // (kstep 0..1, dsub 0..7)
    __shared__ __align__(32) unsigned short ldsP[4 * 1024]; // per-wave P frags

    // ---- preload Q fragments (A layout) straight from global
    bf16x16 aq[4];
    {
        const unsigned short* qrow =
            q_bf + ((size_t)(b * TSEQ + qt * 64 + wave * 16 + nl)) * CDIM
                 + h * DHEAD + g * 8;
#pragma unroll
        for (int ds = 0; ds < 4; ds++) {
            union { bf16x16 v; U4 u[2]; } fr;
            fr.u[0] = *(const U4*)(qrow + ds * 32);        // slots 0..7  (k = 8g+0..7)
            fr.u[1] = *(const U4*)(qrow + ds * 32 + 16);   // slots 8..15 (k = 16+8g+0..7)
            aq[ds] = fr.v;
        }
    }

    floatx8 oacc[8] = {};
    float mrow[8], lrow[8];
#pragma unroll
    for (int r = 0; r < 8; r++) { mrow[r] = -3.0e38f; lrow[r] = 0.0f; }

    const float scale = 0.08838834764831845f;  // 1/sqrt(128)
    const int qabs_wave = PTLEN + qt * 64 + wave * 16;
    const int nck = PTLEN / 64 + qt + 1;       // causal: last chunk containing any key

    for (int ck = 0; ck < nck; ck++) {
        const int c0 = ck * 64;

        // ---- stage K chunk [64 keys x 128 d] into B-fragment layout
        {
            int key = t & 63, dpart = t >> 6;
            const unsigned short* src =
                k_bf + ((size_t)(b * SSEQ + c0 + key)) * CDIM + h * DHEAD + dpart * 64;
#pragma unroll
            for (int c16 = 0; c16 < 4; c16++) {
                int d0 = dpart * 64 + c16 * 16;
                int frag = (d0 >> 5) * 4 + (key >> 4);
                int ln   = (key & 15);
                *(U4*)&ldsK[frag * 512 + ln * 16 + ((d0 & 16) ? 8 : 0)]        = *(const U4*)(src + c16 * 16);
                *(U4*)&ldsK[frag * 512 + (ln + 16) * 16 + ((d0 & 16) ? 8 : 0)] = *(const U4*)(src + c16 * 16 + 8);
            }
        }
        // ---- stage V chunk into B-fragment layout (contraction over keys)
        {
            int dcol = t;  // 0..127
            const unsigned short* src =
                v_bf + ((size_t)(b * SSEQ + c0)) * CDIM + h * DHEAD + dcol;
#pragma unroll
            for (int kgrp = 0; kgrp < 4; kgrp++) {
                union { U4 u[2]; unsigned short s[16]; } tmp;
#pragma unroll
                for (int i = 0; i < 16; i++)
                    tmp.s[i] = src[(size_t)(kgrp * 16 + i) * CDIM];
                int frag = (kgrp >> 1) * 8 + (dcol >> 4);
                int ln   = (dcol & 15);
                *(U4*)&ldsV[frag * 512 + ln * 16 + ((kgrp & 1) ? 8 : 0)]        = tmp.u[0];
                *(U4*)&ldsV[frag * 512 + (ln + 16) * 16 + ((kgrp & 1) ? 8 : 0)] = tmp.u[1];
            }
        }
        __syncthreads();

        // ---- scores = Q @ K^T  (16 WMMAs)
        floatx8 sc[4] = {};
#pragma unroll
        for (int ds = 0; ds < 4; ds++)
#pragma unroll
            for (int ks = 0; ks < 4; ks++) {
                bf16x16 bf = *(const bf16x16*)&ldsK[(ds * 4 + ks) * 512 + lane * 16];
                sc[ks] = wmma_bf16(aq[ds], bf, sc[ks]);
            }

        // ---- scale + causal mask + online softmax (wave-local rows)
        float mnew[8];
#pragma unroll
        for (int r = 0; r < 8; r++) {
            int qabs = qabs_wave + g * 8 + r;
            float mx = mrow[r];
#pragma unroll
            for (int ks = 0; ks < 4; ks++) {
                int keyabs = c0 + ks * 16 + nl;
                float s = sc[ks][r] * scale;
                s = (keyabs <= qabs) ? s : -3.0e38f;
                sc[ks][r] = s;
                mx = fmaxf(mx, s);
            }
#pragma unroll
            for (int off = 1; off < 16; off <<= 1)
                mx = fmaxf(mx, __shfl_xor(mx, off, 32));
            mnew[r] = mx;
        }
#pragma unroll
        for (int r = 0; r < 8; r++) {
            float alpha = __expf(mrow[r] - mnew[r]);
            mrow[r] = mnew[r];
            float rs = 0.0f;
#pragma unroll
            for (int ks = 0; ks < 4; ks++) {
                float p = __expf(sc[ks][r] - mnew[r]);
                sc[ks][r] = p;
                rs += p;
            }
#pragma unroll
            for (int off = 1; off < 16; off <<= 1)
                rs += __shfl_xor(rs, off, 32);
            lrow[r] = lrow[r] * alpha + rs;
#pragma unroll
            for (int j = 0; j < 8; j++) oacc[j][r] *= alpha;
        }

        // ---- restage P through LDS into A-fragment layout (per-wave region)
#pragma unroll
        for (int ks = 0; ks < 4; ks++) {
            int kc   = ks >> 1;
            int slot = (nl & 7) + ((ks & 1) ? 8 : 0);
            int lnof = (nl & 8) ? 16 : 0;
#pragma unroll
            for (int r = 0; r < 8; r++) {
                int m = r + 8 * g;
                ldsP[wave * 1024 + kc * 512 + (m + lnof) * 16 + slot] =
                    f32_to_bf16(sc[ks][r]);
            }
        }
        // ---- O += P @ V  (16 WMMAs)
#pragma unroll
        for (int kc = 0; kc < 2; kc++) {
            bf16x16 pf = *(const bf16x16*)&ldsP[wave * 1024 + kc * 512 + lane * 16];
#pragma unroll
            for (int j = 0; j < 8; j++) {
                bf16x16 vf = *(const bf16x16*)&ldsV[(kc * 8 + j) * 512 + lane * 16];
                oacc[j] = wmma_bf16(pf, vf, oacc[j]);
            }
        }
        __syncthreads();
    }

    // ---- normalize and store attention output (bf16, [B*T, C] head-interleaved)
#pragma unroll
    for (int j = 0; j < 8; j++)
#pragma unroll
        for (int r = 0; r < 8; r++) {
            float val = oacc[j][r] / lrow[r];
            size_t row = (size_t)(b * TSEQ + qt * 64 + wave * 16 + r + 8 * g);
            o_bf[row * CDIM + h * DHEAD + j * 16 + nl] = f32_to_bf16(val);
        }
}

// ---------------------------------------------------------------------------
// Kernel 4: y = yh @ W_o + b_o   (M=4096, N=2048, K=2048; A already bf16)
// ---------------------------------------------------------------------------
__global__ __launch_bounds__(256)
void out_proj(const unsigned short* __restrict__ ybf, const float* __restrict__ Wo,
              const float* __restrict__ bo, float* __restrict__ y)
{
    __shared__ __align__(32) unsigned short ldsA[8 * 512];
    __shared__ __align__(32) unsigned short ldsB[8 * 512];

    const int t = threadIdx.x, wave = t >> 5, lane = t & 31;
    const int m0 = blockIdx.y * 128, n0 = blockIdx.x * 128;

    floatx8 acc[8] = {};

    for (int k0 = 0; k0 < CDIM; k0 += 32) {
        { // stage A (bf16 source)
            int m = t >> 1, half = t & 1;
            const unsigned short* src = ybf + (size_t)(m0 + m) * CDIM + k0 + half * 16;
            union { U4 u[2]; unsigned short s[16]; } ld;
            ld.u[0] = *(const U4*)src;
            ld.u[1] = *(const U4*)(src + 8);
            int sub = m >> 4, lr = m & 15;
#pragma unroll
            for (int i = 0; i < 16; i += 2) {
                int ln   = lr + ((i & 8) ? 16 : 0);
                int slot = (i & 7) + (half ? 8 : 0);
                *(unsigned int*)&ldsA[sub * 512 + ln * 16 + slot] =
                    (unsigned int)ld.s[i] | ((unsigned int)ld.s[i + 1] << 16);
            }
        }
        { // stage B (W_o, f32 -> bf16)
            int n = t & 127, kh = t >> 7;
            const float* src = Wo + (size_t)(k0 + kh * 16) * CDIM + n0 + n;
            unsigned short tmp[16];
#pragma unroll
            for (int i = 0; i < 16; i++) tmp[i] = f32_to_bf16(src[(size_t)i * CDIM]);
            int sub = n >> 4, ln = (n & 15);
#pragma unroll
            for (int i = 0; i < 16; i += 2)
                *(unsigned int*)&ldsB[sub * 512 + (ln + ((i & 8) ? 16 : 0)) * 16 + (i & 7) + kh * 8] =
                    (unsigned int)tmp[i] | ((unsigned int)tmp[i + 1] << 16);
        }
        __syncthreads();

        bf16x16 af = *(const bf16x16*)&ldsA[wave * 512 + lane * 16];
#pragma unroll
        for (int j = 0; j < 8; j++) {
            bf16x16 bf = *(const bf16x16*)&ldsB[j * 512 + lane * 16];
            acc[j] = wmma_bf16(af, bf, acc[j]);
        }
        __syncthreads();
    }

    const int g = lane >> 4, nl = lane & 15;
#pragma unroll
    for (int j = 0; j < 8; j++) {
        int col = n0 + j * 16 + nl;
        float bv = bo[col];
#pragma unroll
        for (int r = 0; r < 8; r++) {
            int row = m0 + wave * 16 + g * 8 + r;
            y[(size_t)row * CDIM + col] = acc[j][r] + bv;
        }
    }
}

// ---------------------------------------------------------------------------
// Launch
// ---------------------------------------------------------------------------
extern "C" void kernel_launch(void* const* d_in, const int* in_sizes, int n_in,
                              void* d_out, int out_size, void* d_ws, size_t ws_size,
                              hipStream_t stream)
{
    (void)in_sizes; (void)n_in; (void)out_size; (void)ws_size;

    const float* x    = (const float*)d_in[0];
    const float* pk   = (const float*)d_in[1];
    const float* pv   = (const float*)d_in[2];
    const float* Wqkv = (const float*)d_in[3];
    const float* bqkv = (const float*)d_in[4];
    const float* Wo   = (const float*)d_in[5];
    const float* bo   = (const float*)d_in[6];

    float* out   = (float*)d_out;
    float* y_out = out;                                        // [B,T,C]
    float* k_out = out + (size_t)BATCH * TSEQ * CDIM;          // [B,S,C]
    float* v_out = k_out + (size_t)BATCH * SSEQ * CDIM;        // [B,S,C]

    char* ws = (char*)d_ws;
    unsigned short* q_bf = (unsigned short*)(ws);                         // 16 MiB
    unsigned short* k_bf = (unsigned short*)(ws + (size_t)16777216);      // 24 MiB
    unsigned short* v_bf = (unsigned short*)(ws + (size_t)41943040);      // 24 MiB
    unsigned short* o_bf = (unsigned short*)(ws + (size_t)67108864);      // 16 MiB

    {
        size_t n = (size_t)BATCH * PTLEN * CDIM;
        copy_past<<<dim3((unsigned)((n + 255) / 256)), dim3(256), 0, stream>>>(
            pk, pv, k_out, v_out, k_bf, v_bf);
    }
    qkv_gemm<<<dim3(N3C / 128, (BATCH * TSEQ) / 128), dim3(256), 0, stream>>>(
        x, Wqkv, bqkv, q_bf, k_out, v_out, k_bf, v_bf);
    attn<<<dim3(TSEQ / 64, NH, BATCH), dim3(128), 0, stream>>>(
        q_bf, k_bf, v_bf, o_bf);
    out_proj<<<dim3(CDIM / 128, (BATCH * TSEQ) / 128), dim3(256), 0, stream>>>(
        o_bf, Wo, bo, y_out);
}
